// Generator_39556648796663
// MI455X (gfx1250) — compile-verified, matrix operand order checked by reading, not run on hardware
//
#include <hip/hip_runtime.h>

// Problem constants (match reference)
#define LSEQ 4096                      // L = 64*64
#define NF   512                       // filters
#define NELEM_K (9 * NF * LSEQ)        // 18874368 kernel elements
#define NDIAG 258                      // distinct Toeplitz diagonals: d/16 in [-2, 255]
#define KH_BYTES ((size_t)NELEM_K * 2) // f16 kernel copy in workspace

typedef _Float16 half4v  __attribute__((ext_vector_type(4)));
typedef _Float16 half8v  __attribute__((ext_vector_type(8)));
typedef _Float16 half16v __attribute__((ext_vector_type(16)));
typedef float    v8f     __attribute__((ext_vector_type(8)));

// ---------------------------------------------------------------------------
// Phase 1: fp32 kernel [c][i][f][s] -> f16 in workspace, applying the causal
// mask at s==0 (keep only strictly-causal channel pairs c > i).
// ---------------------------------------------------------------------------
__global__ __launch_bounds__(256) void prep_kernel_f16(
    const float* __restrict__ k32, _Float16* __restrict__ kh)
{
    int t = blockIdx.x * blockDim.x + threadIdx.x;     // one thread per 4 elems
    if (t >= NELEM_K / 4) return;
    long base = (long)t * 4;
    float4 kv = ((const float4*)k32)[t];
    int s  = (int)(base & (LSEQ - 1));
    int ci = (int)(base >> 21);                        // / (4096*512), 0..8
    int c  = ci / 3;
    int i  = ci - 3 * c;
    float x0 = kv.x;
    if (s == 0 && c <= i) x0 = 0.0f;                   // causal mask on tap 0
    half4v h;
    h[0] = (_Float16)x0;
    h[1] = (_Float16)kv.y;
    h[2] = (_Float16)kv.z;
    h[3] = (_Float16)kv.w;
    ((half4v*)kh)[t] = h;
}

// ---------------------------------------------------------------------------
// Phase 1.5: materialize every distinct Toeplitz B-fragment (32x16 f16) in
// WMMA register layout. Fragment g = (b*3 + c)*NDIAG + di encodes diagonal
// d = (di-2)*16 :  B[k,n] = x[b,c, d + n - k]  (zero when index < 0).
// Layout: 32 lanes x 16 halves, lane-major (32 B per lane) so the consumer
// loads it with two coalesced global_load_b128. 3096 frags = 3.1 MB (L2-hot).
// ---------------------------------------------------------------------------
__global__ __launch_bounds__(256) void build_bfrags(
    const int* __restrict__ example, _Float16* __restrict__ bfr)
{
    const int w    = threadIdx.x >> 5;
    const int lane = threadIdx.x & 31;
    const int g    = blockIdx.x * 8 + w;               // 0..3095 (one frag/wave)
    int b  = g / (3 * NDIAG);
    int r  = g - b * (3 * NDIAG);
    int c  = r / NDIAG;
    int di = r - c * NDIAG;
    int d  = (di - 2) * 16;
    const int n  = lane & 15;
    const int hi = lane >> 4;

    _Float16 vals[16];
    #pragma unroll
    for (int v = 0; v < 8; ++v) {
        int k0 = ((v < 4) ? (2 * v) : (16 + 2 * (v - 4))) + hi * 8;
        #pragma unroll
        for (int h = 0; h < 2; ++h) {
            int idx = d + n - (k0 + h);                // = t' - s, always < LSEQ
            float x = 0.0f;
            if (idx >= 0)
                x = 0.5f * (float)example[((long)b * LSEQ + idx) * 3 + c] - 1.0f;
            vals[2 * v + h] = (_Float16)x;
        }
    }
    _Float16* dst = bfr + (long)g * 512 + lane * 16;
    half8v lo, hv;
    #pragma unroll
    for (int q = 0; q < 8; ++q) { lo[q] = vals[q]; hv[q] = vals[8 + q]; }
    *(half8v*)dst       = lo;
    *(half8v*)(dst + 8) = hv;
}

// ---------------------------------------------------------------------------
// Phase 2: causal Toeplitz GEMM via v_wmma_f32_16x16x32_f16 + fused epilogue.
// Grid: 12 (b,i) x 64 t-blocks. Block: 256 threads = 8 wave32.
// Wave w owns TWO 16-filter tiles (f and f+128) and ALL four 16-wide t-tiles
// of the 64-t block: one A fragment feeds 4 WMMAs, one B fragment feeds 2.
// fc loop = 2 (8 waves x 32 filters x 2 = 512). No LDS in the MMA loop.
// ---------------------------------------------------------------------------
__global__ __launch_bounds__(256) void pixel_conv_wmma(
    const _Float16* __restrict__ kh,        // [c][i][f][s] f16, masked
    const _Float16* __restrict__ bfr,       // Toeplitz fragment pool
    const float*    __restrict__ bias,      // [512]
    const float*    __restrict__ dw,        // [512][4]
    const float*    __restrict__ db,        // [4]
    float*          __restrict__ out)       // [4][4096][3][4]
{
    __shared__ float pacc[64 * 4];          // t-block x C partial outputs
    __shared__ float bias_s[NF];
    __shared__ float dw_s[NF * 4];

    const int tid = threadIdx.x;
    const int bx  = blockIdx.x;
    const int tb  = bx & 63;                // t-block index
    const int bi  = bx >> 6;                // 0..11
    const int b   = bi / 3;
    const int i   = bi - 3 * b;
    const int t0  = tb * 64;

    for (int idx = tid; idx < NF; idx += 256)     bias_s[idx] = bias[idx];
    for (int idx = tid; idx < NF * 4; idx += 256) dw_s[idx]   = dw[idx];
    pacc[tid] = 0.0f;                       // 256 == 64*4 exactly
    __syncthreads();

    const int w    = tid >> 5;              // wave id (wave32)
    const int lane = tid & 31;
    const int n    = lane & 15;
    const int hi   = lane >> 4;
    const int nchunks = 2 * tb + 2;         // causal: s_base <= t0+63

    for (int fc = 0; fc < 2; ++fc) {
        const int fbase = fc * 256;         // wave's two tiles: +w*16 and +128
        const int frowA = fbase + w * 16 + n;
        v8f accA[4] = {v8f{}, v8f{}, v8f{}, v8f{}};
        v8f accB[4] = {v8f{}, v8f{}, v8f{}, v8f{}};

        for (int c = 0; c < 3; ++c) {
            const _Float16* __restrict__ arowA =
                kh + ((long)(c * 3 + i) * NF + frowA) * LSEQ + hi * 8;
            const _Float16* __restrict__ arowB = arowA + (long)128 * LSEQ;
            const _Float16* __restrict__ fgb =
                bfr + ((long)(b * 3 + c) * NDIAG) * 512 + lane * 16;

            for (int sc = 0; sc < nchunks; ++sc) {
                const int sb = sc * 32;
                // Two A fragments (16x32 f16): 2x16B chunks per lane each
                half8v aAlo = *(const half8v*)(arowA + sb);
                half8v aAhi = *(const half8v*)(arowA + sb + 16);
                half8v aBlo = *(const half8v*)(arowB + sb);
                half8v aBhi = *(const half8v*)(arowB + sb + 16);
                half16v aA, aB;
                #pragma unroll
                for (int q = 0; q < 8; ++q) {
                    aA[q] = aAlo[q]; aA[8 + q] = aAhi[q];
                    aB[q] = aBlo[q]; aB[8 + q] = aBhi[q];
                }
                if (sc + 1 < nchunks) {
                    __builtin_prefetch((const void*)(arowA + sb + 32), 0, 1);
                    __builtin_prefetch((const void*)(arowB + sb + 32), 0, 1);
                }

                // Four t-tiles share the A fragments; each pre-swizzled B
                // fragment feeds both filter tiles (2 independent WMMAs).
                #pragma unroll
                for (int tt = 0; tt < 4; ++tt) {
                    const int di = tb * 4 + tt + 2 - 2 * sc;    // diagonal idx
                    const _Float16* bp = fgb + (long)di * 512;
                    half8v blo = *(const half8v*)bp;
                    half8v bhi = *(const half8v*)(bp + 8);
                    half16v bf;
                    #pragma unroll
                    for (int q = 0; q < 8; ++q) { bf[q] = blo[q]; bf[8 + q] = bhi[q]; }
                    accA[tt] = __builtin_amdgcn_wmma_f32_16x16x32_f16(
                        false, aA, false, bf, (short)0, accA[tt], false, false);
                    accB[tt] = __builtin_amdgcn_wmma_f32_16x16x32_f16(
                        false, aB, false, bf, (short)0, accB[tt], false, false);
                }
            }
        }

        // ---- fused epilogue: relu(y + bias) @ dense_w, reduce over this
        //      wave's filter rows in-lane, then LDS atomic accumulate ----
        #pragma unroll
        for (int tt = 0; tt < 4; ++tt) {
            float p0 = 0.f, p1 = 0.f, p2 = 0.f, p3 = 0.f;
            #pragma unroll
            for (int r = 0; r < 8; ++r) {
                int m  = r + 8 * hi;                   // C/D row per ISA layout
                int fA = fbase + w * 16 + m;
                int fB = fA + 128;
                float vA = fmaxf(accA[tt][r] + bias_s[fA], 0.0f);
                float vB = fmaxf(accB[tt][r] + bias_s[fB], 0.0f);
                p0 += vA * dw_s[fA * 4 + 0] + vB * dw_s[fB * 4 + 0];
                p1 += vA * dw_s[fA * 4 + 1] + vB * dw_s[fB * 4 + 1];
                p2 += vA * dw_s[fA * 4 + 2] + vB * dw_s[fB * 4 + 2];
                p3 += vA * dw_s[fA * 4 + 3] + vB * dw_s[fB * 4 + 3];
            }
            int tl = tt * 16 + n;
            atomicAdd(&pacc[tl * 4 + 0], p0);
            atomicAdd(&pacc[tl * 4 + 1], p1);
            atomicAdd(&pacc[tl * 4 + 2], p2);
            atomicAdd(&pacc[tl * 4 + 3], p3);
        }
    }
    __syncthreads();

    // ---- final store: [b][t][i][co] = partial + dense_b (written once) ----
    {
        int tl = tid >> 2;
        int co = tid & 3;
        out[(((long)b * LSEQ + (t0 + tl)) * 3 + i) * 4 + co] = pacc[tid] + db[co];
    }
}

// ---------------------------------------------------------------------------
extern "C" void kernel_launch(void* const* d_in, const int* in_sizes, int n_in,
                              void* d_out, int out_size, void* d_ws, size_t ws_size,
                              hipStream_t stream)
{
    (void)in_sizes; (void)n_in; (void)out_size; (void)ws_size;
    const int*   example = (const int*)  d_in[0];   // [4,64,64,3] int32
    const float* k32     = (const float*)d_in[1];   // [3,3,512,4096] f32
    const float* bias    = (const float*)d_in[2];   // [512]
    const float* dw      = (const float*)d_in[3];   // [512,4]
    const float* db      = (const float*)d_in[4];   // [4]
    float*       out     = (float*)d_out;           // [4,4096,3,4] f32

    _Float16* kh  = (_Float16*)d_ws;                        // 37.75 MB
    _Float16* bfr = (_Float16*)((char*)d_ws + KH_BYTES);    // 3.1 MB frag pool

    prep_kernel_f16<<<NELEM_K / 4 / 256, 256, 0, stream>>>(k32, kh);
    build_bfrags<<<(4 * 3 * NDIAG) / 8, 256, 0, stream>>>(example, bfr);
    pixel_conv_wmma<<<12 * 64, 256, 0, stream>>>(kh, bfr, bias, dw, db, out);
}
